// CPELayer_ResAG_19335942766951
// MI455X (gfx1250) — matrix operand order repaired and checked by reading, not compiled, not run on hardware
//
#include <hip/hip_runtime.h>

typedef __attribute__((ext_vector_type(16))) _Float16 v16h;
typedef __attribute__((ext_vector_type(8)))  float    v8f;

#define BT    4928      // b*t = 64*77
#define DIN   768
#define DOUT  1280
#define RANKK 4
#define SCALEF 0.25f    // ALPHA/RANK * MULTIPLIER

// ---------------------------------------------------------------------------
// Kernel 1: per-token LoRA "down" projection.
// One wave32 per token: down[h] = sum_d ed[token][d] * lora_down[c][d][h]
// ---------------------------------------------------------------------------
__global__ __launch_bounds__(256) void lora_down_kernel(
    const float* __restrict__ ed, const int* __restrict__ cidx,
    const float* __restrict__ lora_down, float* __restrict__ down_ws) {
  const int gtid  = blockIdx.x * blockDim.x + threadIdx.x;
  const int token = gtid >> 5;
  const int lane  = threadIdx.x & 31;
  if (token >= BT) return;

  const int c = cidx[token];
  const float* __restrict__ edp = ed + (size_t)token * DIN;
  const float* __restrict__ ldp = lora_down + (size_t)c * (DIN * RANKK);

  float p0 = 0.f, p1 = 0.f, p2 = 0.f, p3 = 0.f;
#pragma unroll 4
  for (int d = lane; d < DIN; d += 32) {
    const float e = edp[d];
    const float4 w = *(const float4*)(ldp + d * 4);   // lora_down[c][d][0..3]
    p0 += e * w.x; p1 += e * w.y; p2 += e * w.z; p3 += e * w.w;
  }
#pragma unroll
  for (int off = 16; off > 0; off >>= 1) {
    p0 += __shfl_xor(p0, off, 32);
    p1 += __shfl_xor(p1, off, 32);
    p2 += __shfl_xor(p2, off, 32);
    p3 += __shfl_xor(p3, off, 32);
  }
  if (lane == 0) {
    float4 r; r.x = p0; r.y = p1; r.z = p2; r.w = p3;
    *(float4*)(down_ws + (size_t)token * 4) = r;
  }
}

// ---------------------------------------------------------------------------
// Fragment loaders (f32 source -> f16 fragment, ISA 16-bit layouts)
// A (16x32): lane m = L&15, elems 0-7 = K[k0+h*8..], 8-15 = K[k0+16+h*8..]
// B (32x16): lane n = L&15, elems 0-7 = K[k0+h*8..], 8-15 = K[k0+16+h*8..]
// ---------------------------------------------------------------------------
__device__ __forceinline__ v16h load_a_frag(const float* __restrict__ row,
                                            int k0, int half) {
  const float* ap = row + k0 + (half << 3);
  const float4 f0 = ((const float4*)ap)[0];
  const float4 f1 = ((const float4*)ap)[1];
  const float4 f2 = ((const float4*)(ap + 16))[0];
  const float4 f3 = ((const float4*)(ap + 16))[1];
  v16h a;
  a[0]=(_Float16)f0.x; a[1]=(_Float16)f0.y; a[2]=(_Float16)f0.z; a[3]=(_Float16)f0.w;
  a[4]=(_Float16)f1.x; a[5]=(_Float16)f1.y; a[6]=(_Float16)f1.z; a[7]=(_Float16)f1.w;
  a[8]=(_Float16)f2.x; a[9]=(_Float16)f2.y; a[10]=(_Float16)f2.z; a[11]=(_Float16)f2.w;
  a[12]=(_Float16)f3.x; a[13]=(_Float16)f3.y; a[14]=(_Float16)f3.z; a[15]=(_Float16)f3.w;
  return a;
}

__device__ __forceinline__ v16h load_b_frag(const float* __restrict__ row,
                                            int k0, int half) {
  const float* bp = row + k0 + (half << 3);
  const float4 g0 = ((const float4*)bp)[0];
  const float4 g1 = ((const float4*)bp)[1];
  const float4 g2 = ((const float4*)(bp + 16))[0];
  const float4 g3 = ((const float4*)(bp + 16))[1];
  v16h b;
  b[0]=(_Float16)g0.x; b[1]=(_Float16)g0.y; b[2]=(_Float16)g0.z; b[3]=(_Float16)g0.w;
  b[4]=(_Float16)g1.x; b[5]=(_Float16)g1.y; b[6]=(_Float16)g1.z; b[7]=(_Float16)g1.w;
  b[8]=(_Float16)g2.x; b[9]=(_Float16)g2.y; b[10]=(_Float16)g2.z; b[11]=(_Float16)g2.w;
  b[12]=(_Float16)g3.x; b[13]=(_Float16)g3.y; b[14]=(_Float16)g3.z; b[15]=(_Float16)g3.w;
  return b;
}

// ---------------------------------------------------------------------------
// Kernel 2: out = x @ W^T + b_lin + SCALE * (down @ lora_up[c]), via
// v_wmma_f32_16x16x32_f16 with f32 accumulation.
// Block = 256 threads = 8 waves arranged 2(M) x 4(N); each wave computes a
// 32x64 output strip (8 accumulators: 2 M-subtiles x 4 N-subtiles), so each
// B fragment feeds 2 WMMAs and each A fragment feeds 4. Block tile = 64x256.
// ---------------------------------------------------------------------------
__global__ __launch_bounds__(256) void gemm_lora_kernel(
    const float* __restrict__ x, const int* __restrict__ cidx,
    const float* __restrict__ lora_up, const float* __restrict__ Wm,
    const float* __restrict__ b_lin, const float* __restrict__ down_ws,
    float* __restrict__ out) {
  const int wid  = threadIdx.x >> 5;
  const int lane = threadIdx.x & 31;
  const int half = lane >> 4;      // 0 or 1
  const int l15  = lane & 15;

  const int row0 = blockIdx.y * 64  + (wid & 1) * 32;   // M strip origin
  const int col0 = blockIdx.x * 256 + (wid >> 1) * 64;  // N strip origin

  v8f acc[2][4] = {};                                   // [m-subtile][n-subtile]

  const float* __restrict__ arow0 = x + (size_t)(row0 + l15) * DIN;
  const float* __restrict__ arow1 = arow0 + (size_t)16 * DIN;
  const float* __restrict__ brow  = Wm + (size_t)(col0 + l15) * DIN;

  for (int k0 = 0; k0 < DIN; k0 += 32) {
    // Speculative prefetch of next K chunk (overshoot past row end is
    // silently dropped per ISA prefetch semantics) — no branch.
    __builtin_prefetch(arow0 + k0 + 32, 0, 1);
    __builtin_prefetch(arow1 + k0 + 32, 0, 1);
    __builtin_prefetch(brow  + k0 + 32, 0, 1);

    const v16h a0 = load_a_frag(arow0, k0, half);
    const v16h a1 = load_a_frag(arow1, k0, half);

#pragma unroll
    for (int sub = 0; sub < 4; ++sub) {
      const v16h b = load_b_frag(brow + (size_t)(sub * 16) * DIN, k0, half);
      acc[0][sub] = __builtin_amdgcn_wmma_f32_16x16x32_f16(
          false, a0, false, b, (short)0, acc[0][sub], false, false);
      acc[1][sub] = __builtin_amdgcn_wmma_f32_16x16x32_f16(
          false, a1, false, b, (short)0, acc[1][sub], false, false);
    }
  }

  // ---- Epilogue: C/D layout — VGPR r: M = r + 8*half, N = l15 per 16-col sub
#pragma unroll
  for (int mt = 0; mt < 2; ++mt) {
#pragma unroll
    for (int r = 0; r < 8; ++r) {
      const int m = row0 + mt * 16 + r + (half << 3);
      const int c = cidx[m];
      const float4 dn = *(const float4*)(down_ws + (size_t)m * 4);
      const float* __restrict__ up = lora_up + (size_t)c * (RANKK * DOUT);
#pragma unroll
      for (int sub = 0; sub < 4; ++sub) {
        const int n = col0 + sub * 16 + l15;
        const float u = dn.x * up[n] + dn.y * up[DOUT + n] +
                        dn.z * up[2 * DOUT + n] + dn.w * up[3 * DOUT + n];
        out[(size_t)m * DOUT + n] = acc[mt][sub][r] + b_lin[n] + SCALEF * u;
      }
    }
  }
}

// ---------------------------------------------------------------------------
extern "C" void kernel_launch(void* const* d_in, const int* in_sizes, int n_in,
                              void* d_out, int out_size, void* d_ws, size_t ws_size,
                              hipStream_t stream) {
  const float* x    = (const float*)d_in[0];
  const float* ed   = (const float*)d_in[1];
  const int*   cidx = (const int*)  d_in[2];
  const float* ldn  = (const float*)d_in[3];
  const float* lup  = (const float*)d_in[4];
  const float* Wm   = (const float*)d_in[5];
  const float* bl   = (const float*)d_in[6];
  float* out = (float*)d_out;
  float* down_ws = (float*)d_ws;            // BT*4 floats = 78,848 bytes

  // Kernel 1: one wave per token -> 4928 waves = 616 blocks of 256
  lora_down_kernel<<<dim3((BT * 32 + 255) / 256), 256, 0, stream>>>(
      ed, cidx, ldn, down_ws);

  // Kernel 2: grid = (N blocks, M blocks) = (1280/256, 4928/64) = (5, 77)
  gemm_lora_kernel<<<dim3(DOUT / 256, BT / 64), 256, 0, stream>>>(
      x, cidx, lup, Wm, bl, down_ws, out);
}
